// BlockDiffusionDecoder_39075612459671
// MI455X (gfx1250) — compile-verified, matrix operand order checked by reading, not run on hardware
//
#include <hip/hip_runtime.h>
#include <math.h>

// ---------------------------------------------------------------------------
// MI455X (gfx1250) implementation. Compute-bound (~530 GF vs ~125MB working
// set that lives in the 192MB L2) -> everything matrix-shaped runs on
// v_wmma_f32_16x16x32_bf16 (bf16 inputs, f32 accumulate), wave32, WGP LDS
// staging. Elementwise work (LN, adaLN modulation, RoPE, softmax, GELU,
// residual+gate) is fused into GEMM epilogues / small kernels.
// ---------------------------------------------------------------------------

typedef __bf16 bf16;
typedef __attribute__((ext_vector_type(16))) __bf16 v16bf;
typedef __attribute__((ext_vector_type(8)))  __bf16 bf16x8;
typedef __attribute__((ext_vector_type(8)))  float  v8f;

// gfx1250 async global->LDS copy path (ASYNCcnt), if the toolchain exposes it.
// Probe result: param0 is int __attribute__((vector_size(16))) in AS1 (global),
// param1 is the LDS-side pointer, then (imm offset, cpol).
#if defined(__gfx1250__) && __has_builtin(__builtin_amdgcn_global_load_async_to_lds_b128) && __has_builtin(__builtin_amdgcn_s_wait_asynccnt)
#define USE_ASYNC_LDS 1
typedef int v4i_ __attribute__((vector_size(16)));
typedef __attribute__((address_space(1))) v4i_ g_v4i;
typedef __attribute__((address_space(3))) v4i_ l_v4i;
#endif

__device__ __forceinline__ v8f vzero8() {
    v8f z;
#pragma unroll
    for (int i = 0; i < 8; ++i) z[i] = 0.f;
    return z;
}

// CDNA5 16-bit A/B fragment (16x32): lane L<16 holds row (L), K in
// {kb..kb+7, kb+16..kb+23} with kb = (L>=16)*8 -> two 16-byte chunks.
__device__ __forceinline__ v16bf load_frag16(const bf16* base, int stride, int lane) {
    const bf16* p = base + (lane & 15) * stride + ((lane >> 4) << 3);
    bf16x8 lo = *(const bf16x8*)p;
    bf16x8 hi = *(const bf16x8*)(p + 16);
    v16bf v;
#pragma unroll
    for (int i = 0; i < 8; ++i) { v[i] = lo[i]; v[i + 8] = hi[i]; }
    return v;
}

__device__ __forceinline__ v8f wmma_bf16(v16bf a, v16bf b, v8f c) {
    return __builtin_amdgcn_wmma_f32_16x16x32_bf16(false, a, false, b, (short)0, c, false, false);
}

// ---------------------------------------------------------------------------
// 1) adaLN modulation: mod[b, 0:6144] = c[b, 0:128] @ adaLN_w + adaLN_b
// ---------------------------------------------------------------------------
__global__ __launch_bounds__(256) void mod_kernel(const float* __restrict__ c,
                                                  const float* __restrict__ W,
                                                  const float* __restrict__ bias,
                                                  float* __restrict__ mod) {
    __shared__ float cs[128];
    int b = blockIdx.x / 24;
    int n = (blockIdx.x % 24) * 256 + threadIdx.x;
    if (threadIdx.x < 128) cs[threadIdx.x] = c[b * 128 + threadIdx.x];
    __syncthreads();
    float acc = bias[n];
#pragma unroll 8
    for (int k = 0; k < 128; ++k) acc += cs[k] * W[(size_t)k * 6144 + n];
    mod[(size_t)b * 6144 + n] = acc;
}

// ---------------------------------------------------------------------------
// 2) LayerNorm (+ optional adaLN shift/scale), f32 in -> bf16 out.
//    One block per row (D=1024, 4 elems/thread).
// ---------------------------------------------------------------------------
__global__ __launch_bounds__(256) void ln_kernel(const float* __restrict__ x,
                                                 const float* __restrict__ w,
                                                 const float* __restrict__ mod,
                                                 int shOff, int scOff,
                                                 bf16* __restrict__ out) {
    __shared__ float red1[8], red2[8];
    int row = blockIdx.x, b = row >> 8, tid = threadIdx.x;
    int lane = tid & 31, wv = tid >> 5;
    const float* xr = x + (size_t)row * 1024;
    float v[4];
#pragma unroll
    for (int i = 0; i < 4; ++i) v[i] = xr[tid + i * 256];
    float s = v[0] + v[1] + v[2] + v[3];
#pragma unroll
    for (int off = 16; off; off >>= 1) s += __shfl_xor(s, off, 32);
    if (lane == 0) red1[wv] = s;
    __syncthreads();
    float tot = 0.f;
#pragma unroll
    for (int i = 0; i < 8; ++i) tot += red1[i];
    float mean = tot * (1.f / 1024.f);
    float q = 0.f;
#pragma unroll
    for (int i = 0; i < 4; ++i) { float d = v[i] - mean; q += d * d; }
#pragma unroll
    for (int off = 16; off; off >>= 1) q += __shfl_xor(q, off, 32);
    if (lane == 0) red2[wv] = q;
    __syncthreads();
    float qt = 0.f;
#pragma unroll
    for (int i = 0; i < 8; ++i) qt += red2[i];
    float rstd = rsqrtf(qt * (1.f / 1024.f) + 1e-5f);
#pragma unroll
    for (int i = 0; i < 4; ++i) {
        int d = tid + i * 256;
        float y = (v[i] - mean) * rstd * w[d];
        if (mod) {
            const float* mb = mod + (size_t)b * 6144;
            y = y * (1.f + mb[scOff + d]) + mb[shOff + d];
        }
        out[(size_t)row * 1024 + d] = (bf16)y;
    }
}

// ---------------------------------------------------------------------------
// 3) WMMA GEMM: C[M,N] = A[M,K](bf16 or f32->bf16) * B[K,N](f32->bf16)
//    128x256x32 block tile, 8 waves as 2x4, each wave 64x64 (4x4 WMMAs).
//    Weight tile converted & transposed into LDS with packed b32 stores.
//    Fused epilogues.
// ---------------------------------------------------------------------------
enum { EPI_BF16 = 0, EPI_BIAS_GELU = 1, EPI_RES_GATE = 2, EPI_RES = 3, EPI_RES_GATE_BIAS = 4 };

template <int EPI, bool AF32>
__global__ __launch_bounds__(256) void gemm_kernel(const void* __restrict__ Av,
                                                   const float* __restrict__ Bw,
                                                   int M, int N, int K,
                                                   bf16* __restrict__ outB,
                                                   float* __restrict__ outF,
                                                   const float* __restrict__ res,
                                                   const float* __restrict__ bias,
                                                   const float* __restrict__ mod,
                                                   int gateOff) {
    constexpr int LDA = 48, LDB = 48;  // padded strides, 96B (16B aligned)
    __shared__ __align__(16) bf16 As[128 * LDA];
    __shared__ __align__(16) bf16 Bs[256 * LDB];

    int tid = threadIdx.x, lane = tid & 31, w = tid >> 5;
    int wm = w >> 2, wn = w & 3;
    int m0 = blockIdx.y * 128, n0 = blockIdx.x * 256;

    v8f acc[4][4];
#pragma unroll
    for (int f = 0; f < 4; ++f)
#pragma unroll
        for (int g = 0; g < 4; ++g) acc[f][g] = vzero8();

    const bf16* A16 = (const bf16*)Av;
    const float* A32 = (const float*)Av;
    int arow = tid >> 1, ahalf = tid & 1;   // A: half row per thread (16 elems)
    int bn0 = (tid & 15) * 16;              // B: 16 consecutive n ...
    int bkp = (tid >> 4) * 2;               //    ... for a pair of k rows

    for (int k0 = 0; k0 < K; k0 += 32) {
        __syncthreads();
        // ---- stage A tile (128 x 32 bf16) ----
        bf16* adst = &As[arow * LDA + ahalf * 16];
        if (AF32) {
            const float* src = A32 + (size_t)(m0 + arow) * K + k0 + ahalf * 16;
            bf16x8 p0, p1;
#pragma unroll
            for (int i = 0; i < 8; ++i) { p0[i] = (bf16)src[i]; p1[i] = (bf16)src[8 + i]; }
            *(bf16x8*)adst = p0;
            *(bf16x8*)(adst + 8) = p1;
        } else {
            const bf16* src = A16 + (size_t)(m0 + arow) * K + k0 + ahalf * 16;
#ifdef USE_ASYNC_LDS
            __builtin_amdgcn_global_load_async_to_lds_b128((g_v4i*)src, (l_v4i*)adst, 0, 0);
            __builtin_amdgcn_global_load_async_to_lds_b128((g_v4i*)(src + 8), (l_v4i*)(adst + 8), 0, 0);
#else
            *(bf16x8*)adst = *(const bf16x8*)src;
            *(bf16x8*)(adst + 8) = *(const bf16x8*)(src + 8);
#endif
        }
        // ---- stage B tile (32 x 256 f32 -> bf16, transposed, packed b32) ----
        {
            const float* b0 = Bw + (size_t)(k0 + bkp) * N + n0 + bn0;
            const float* b1 = b0 + N;
            if (k0 + 32 < K) {  // prefetch next K-step's weight rows into L2
                __builtin_prefetch(b0 + (size_t)32 * N, 0, 1);
                __builtin_prefetch(b1 + (size_t)32 * N, 0, 1);
            }
#pragma unroll
            for (int i = 0; i < 16; ++i) {
                union { unsigned u; bf16 h[2]; } pk;
                pk.h[0] = (bf16)b0[i];
                pk.h[1] = (bf16)b1[i];
                *(unsigned*)&Bs[(bn0 + i) * LDB + bkp] = pk.u;
            }
        }
#ifdef USE_ASYNC_LDS
        if (!AF32) __builtin_amdgcn_s_wait_asynccnt(0);
#endif
        __syncthreads();

        v16bf af[4], bfr[4];
#pragma unroll
        for (int f = 0; f < 4; ++f) af[f] = load_frag16(&As[(wm * 64 + f * 16) * LDA], LDA, lane);
#pragma unroll
        for (int g = 0; g < 4; ++g) bfr[g] = load_frag16(&Bs[(wn * 64 + g * 16) * LDB], LDB, lane);
#pragma unroll
        for (int f = 0; f < 4; ++f)
#pragma unroll
            for (int g = 0; g < 4; ++g) acc[f][g] = wmma_bf16(af[f], bfr[g], acc[f][g]);
    }

    int mBase = m0 + wm * 64 + ((lane >> 4) << 3);
    int nBase = n0 + wn * 64 + (lane & 15);
#pragma unroll
    for (int f = 0; f < 4; ++f)
#pragma unroll
        for (int g = 0; g < 4; ++g) {
            int ncol = nBase + g * 16;
#pragma unroll
            for (int j = 0; j < 8; ++j) {
                int m = mBase + f * 16 + j;
                size_t o = (size_t)m * N + ncol;
                float v = acc[f][g][j];
                if constexpr (EPI == EPI_BF16) {
                    outB[o] = (bf16)v;
                } else if constexpr (EPI == EPI_BIAS_GELU) {
                    float t = v + bias[ncol];
                    float gl = 0.5f * t * (1.f + tanhf(0.7978845608f * (t + 0.044715f * t * t * t)));
                    outB[o] = (bf16)gl;
                } else if constexpr (EPI == EPI_RES_GATE) {
                    outF[o] = res[o] + mod[(size_t)(m >> 8) * 6144 + gateOff + ncol] * v;
                } else if constexpr (EPI == EPI_RES) {
                    outF[o] = res[o] + v;
                } else {  // EPI_RES_GATE_BIAS
                    outF[o] = res[o] + mod[(size_t)(m >> 8) * 6144 + gateOff + ncol] * (v + bias[ncol]);
                }
            }
        }
}

// ---------------------------------------------------------------------------
// 4) RoPE in place on qkv (B, 2S, 3, H, 64) bf16; thread owns a (d, d+32) pair.
// ---------------------------------------------------------------------------
__global__ __launch_bounds__(256) void rope_kernel(bf16* __restrict__ qkv,
                                                   const float* __restrict__ cosp,
                                                   const float* __restrict__ sinp) {
    int idx = blockIdx.x * 256 + threadIdx.x;   // 16,777,216 pairs
    int d = idx & 31;
    int h = (idx >> 5) & 15;
    int which = (idx >> 9) & 1;                 // 0=q, 1=k
    int s = (idx >> 10) & 255;
    int b = idx >> 18;
    int pos = s & 127;                          // per-half positions
    size_t base = ((size_t)b * 256 + s) * 3072 + which * 1024 + h * 64;
    float c0 = cosp[pos * 64 + d],      s0 = sinp[pos * 64 + d];
    float c1 = cosp[pos * 64 + d + 32], s1 = sinp[pos * 64 + d + 32];
    float x0 = (float)qkv[base + d];
    float x1 = (float)qkv[base + d + 32];
    qkv[base + d]      = (bf16)(x0 * c0 - x1 * s0);
    qkv[base + d + 32] = (bf16)(x1 * c1 + x0 * s1);
}

// ---------------------------------------------------------------------------
// 5) Flash-style WMMA attention. One block = one (b,h). 8 waves, each owns
//    two 16-row q tiles. K chunk (256x64) and V^T chunk (64x256) in LDS,
//    S in registers, per-wave P tile bounced through LDS for re-layout.
//    MASKMODE 0: 2D token mask [256,256]; 1: per-(b,col) encoder mask.
// ---------------------------------------------------------------------------
template <int MASKMODE>
__global__ __launch_bounds__(256) void attn_kernel(const bf16* __restrict__ qp,
                                                   const bf16* __restrict__ kp,
                                                   const bf16* __restrict__ vp,
                                                   long qB, long kB, int qStride, int kStride,
                                                   int kvLen,
                                                   const unsigned char* __restrict__ maskp,
                                                   bf16* __restrict__ outp, long outB, int outStride,
                                                   int H) {
    __shared__ __align__(16) bf16 Ks[256 * 72];      // K chunk, row-major (pad->72)
    __shared__ __align__(16) bf16 Vt[64 * 264];      // V chunk transposed [feat][key]
    __shared__ __align__(16) bf16 Ps[8 * 16 * 264];  // per-wave P tile [16][264]

    int bh = blockIdx.x, b = bh / H, h = bh % H;
    const bf16* Q = qp + (long)b * qB + h * 64;
    const bf16* K = kp + (long)b * kB + h * 64;
    const bf16* V = vp + (long)b * kB + h * 64;
    bf16* O = outp + (long)b * outB + h * 64;

    int tid = threadIdx.x, lane = tid & 31, w = tid >> 5;
    int rowOff = (lane >> 4) << 3;
    int colL = lane & 15;
    bf16* P = &Ps[w * 16 * 264];

    float mrow[2][8], srow[2][8];
    v8f ao[2][4];
#pragma unroll
    for (int t = 0; t < 2; ++t) {
#pragma unroll
        for (int j = 0; j < 8; ++j) { mrow[t][j] = -1e30f; srow[t][j] = 0.f; }
#pragma unroll
        for (int ft = 0; ft < 4; ++ft) ao[t][ft] = vzero8();
    }

    int nchunks = kvLen >> 8;
    for (int ch = 0; ch < nchunks; ++ch) {
        __syncthreads();
        {   // cooperative chunk stage: thread t owns key row t
            const bf16* ks = K + (long)(ch * 256 + tid) * kStride;
            bf16* kd = &Ks[tid * 72];
#ifdef USE_ASYNC_LDS
#pragma unroll
            for (int i = 0; i < 8; ++i)
                __builtin_amdgcn_global_load_async_to_lds_b128((g_v4i*)(ks + i * 8),
                                                               (l_v4i*)(kd + i * 8), 0, 0);
#else
#pragma unroll
            for (int i = 0; i < 8; ++i) ((bf16x8*)kd)[i] = ((const bf16x8*)ks)[i];
#endif
            const bf16* vs = V + (long)(ch * 256 + tid) * kStride;
#pragma unroll
            for (int d = 0; d < 64; ++d) Vt[d * 264 + tid] = vs[d];
        }
#ifdef USE_ASYNC_LDS
        __builtin_amdgcn_s_wait_asynccnt(0);
#endif
        __syncthreads();

#pragma unroll
        for (int t = 0; t < 2; ++t) {
            int q0 = (w + t * 8) * 16;
            v16bf qf0 = load_frag16(Q + (long)q0 * qStride, qStride, lane);
            v16bf qf1 = load_frag16(Q + (long)q0 * qStride + 32, qStride, lane);

            v8f sacc[16];
#pragma unroll
            for (int nt = 0; nt < 16; ++nt) {
                v8f s = vzero8();
                s = wmma_bf16(qf0, load_frag16(&Ks[nt * 16 * 72], 72, lane), s);
                s = wmma_bf16(qf1, load_frag16(&Ks[nt * 16 * 72 + 32], 72, lane), s);
                sacc[nt] = s;
            }
            // scale + mask
#pragma unroll
            for (int nt = 0; nt < 16; ++nt) {
                int col = ch * 256 + nt * 16 + colL;
#pragma unroll
                for (int j = 0; j < 8; ++j) {
                    float sv = sacc[nt][j] * 0.125f;  // 1/sqrt(64)
                    bool ok;
                    if (MASKMODE == 0) ok = maskp[(q0 + j + rowOff) * 256 + col] != 0;
                    else               ok = maskp[(long)b * kvLen + col] != 0;
                    sacc[nt][j] = ok ? sv : -1e30f;
                }
            }
            // online softmax update
            float nm[8], rs[8];
#pragma unroll
            for (int j = 0; j < 8; ++j) {
                float mx = sacc[0][j];
#pragma unroll
                for (int nt = 1; nt < 16; ++nt) mx = fmaxf(mx, sacc[nt][j]);
#pragma unroll
                for (int off = 1; off < 16; off <<= 1) mx = fmaxf(mx, __shfl_xor(mx, off, 32));
                nm[j] = fmaxf(mrow[t][j], mx);
                float f = __expf(mrow[t][j] - nm[j]);
                mrow[t][j] = nm[j];
                srow[t][j] *= f;
#pragma unroll
                for (int ft = 0; ft < 4; ++ft) ao[t][ft][j] *= f;
                rs[j] = 0.f;
            }
#pragma unroll
            for (int nt = 0; nt < 16; ++nt)
#pragma unroll
                for (int j = 0; j < 8; ++j) {
                    float p = __expf(sacc[nt][j] - nm[j]);
                    rs[j] += p;
                    P[(j + rowOff) * 264 + nt * 16 + colL] = (bf16)p;
                }
#pragma unroll
            for (int j = 0; j < 8; ++j) {
                float s = rs[j];
#pragma unroll
                for (int off = 1; off < 16; off <<= 1) s += __shfl_xor(s, off, 32);
                srow[t][j] += s;
            }
            // O += P @ V  (A frags from P LDS, B frags from V^T LDS)
#pragma unroll
            for (int ks2 = 0; ks2 < 8; ++ks2) {
                v16bf pf = load_frag16(P + ks2 * 32, 264, lane);
#pragma unroll
                for (int ft = 0; ft < 4; ++ft)
                    ao[t][ft] = wmma_bf16(pf, load_frag16(&Vt[ft * 16 * 264 + ks2 * 32], 264, lane),
                                          ao[t][ft]);
            }
        }
    }
    // normalize + store (b, s, h, d) bf16
#pragma unroll
    for (int t = 0; t < 2; ++t) {
        int q0 = (w + t * 8) * 16;
#pragma unroll
        for (int ft = 0; ft < 4; ++ft)
#pragma unroll
            for (int j = 0; j < 8; ++j) {
                int m = q0 + j + rowOff;
                O[(long)m * outStride + ft * 16 + colL] = (bf16)(ao[t][ft][j] / srow[t][j]);
            }
    }
}

// ---------------------------------------------------------------------------
// Launch: full block. d_out doubles as the running f32 residual stream.
// Workspace arenas (194 MiB total):
//   [0,2MiB)    mod (64x6144 f32)
//   [2,34MiB)   X arena  (bf16, 16.7M): xn -> xc -> ac
//   [34,66MiB)  A arena  (bf16, 16.7M): a -> qc -> h
//   [66,194MiB) BIG arena(bf16, 67.1M): qkv -> kvc -> mlp hidden
// ---------------------------------------------------------------------------
extern "C" void kernel_launch(void* const* d_in, const int* in_sizes, int n_in,
                              void* d_out, int out_size, void* d_ws, size_t ws_size,
                              hipStream_t stream) {
    (void)in_sizes; (void)n_in; (void)out_size; (void)ws_size;
    const float* x      = (const float*)d_in[0];
    const float* c      = (const float*)d_in[1];
    const float* enc    = (const float*)d_in[2];
    const unsigned char* encMask = (const unsigned char*)d_in[3];
    const unsigned char* mask2d  = (const unsigned char*)d_in[4];
    const float* cosp   = (const float*)d_in[5];
    const float* sinp   = (const float*)d_in[6];
    const float* norm1w = (const float*)d_in[7];
    const float* w_qkv  = (const float*)d_in[8];
    const float* w_ao   = (const float*)d_in[9];
    const float* adaW   = (const float*)d_in[10];
    const float* adaB   = (const float*)d_in[11];
    const float* canw   = (const float*)d_in[12];
    const float* w_q    = (const float*)d_in[13];
    const float* w_kv   = (const float*)d_in[14];
    const float* w_o    = (const float*)d_in[15];
    const float* norm2w = (const float*)d_in[16];
    const float* w_m1   = (const float*)d_in[17];
    const float* b_m1   = (const float*)d_in[18];
    const float* w_m2   = (const float*)d_in[19];
    const float* b_m2   = (const float*)d_in[20];
    float* out = (float*)d_out;

    char* ws = (char*)d_ws;
    float* mod = (float*)ws;
    bf16* Xb  = (bf16*)(ws + (size_t)2  * 1024 * 1024);
    bf16* Ab  = (bf16*)(ws + (size_t)34 * 1024 * 1024);
    bf16* BIG = (bf16*)(ws + (size_t)66 * 1024 * 1024);

    // adaLN chunk offsets within mod[b, 6*1024]
    const int SH_MSA = 0, SC_MSA = 1024, G_MSA = 2048, SH_MLP = 3072, SC_MLP = 4096, G_MLP = 5120;

    // 1) modulation
    mod_kernel<<<1536, 256, 0, stream>>>(c, adaW, adaB, mod);
    // 2) xn = LN1(x)*(1+sc)+sh
    ln_kernel<<<16384, 256, 0, stream>>>(x, norm1w, mod, SH_MSA, SC_MSA, Xb);
    // 3) qkv = xn @ w_qkv   (16384 x 3072 x 1024)
    gemm_kernel<EPI_BF16, false><<<dim3(12, 128), 256, 0, stream>>>(
        Xb, w_qkv, 16384, 3072, 1024, BIG, nullptr, nullptr, nullptr, nullptr, 0);
    // 4) RoPE on q,k in place
    rope_kernel<<<65536, 256, 0, stream>>>(BIG, cosp, sinp);
    // 5) self-attention (block mask) -> Ab
    attn_kernel<0><<<1024, 256, 0, stream>>>(
        BIG, BIG + 1024, BIG + 2048, 786432L, 786432L, 3072, 3072, 256,
        mask2d, Ab, 262144L, 1024, 16);
    // 6) x = x + g_msa * (a @ w_attn_out)
    gemm_kernel<EPI_RES_GATE, false><<<dim3(4, 128), 256, 0, stream>>>(
        Ab, w_ao, 16384, 1024, 1024, nullptr, out, x, nullptr, mod, G_MSA);
    // 7) xc = LN_ca(x)
    ln_kernel<<<16384, 256, 0, stream>>>(out, canw, nullptr, 0, 0, Xb);
    // 8) qc = xc @ w_q
    gemm_kernel<EPI_BF16, false><<<dim3(4, 128), 256, 0, stream>>>(
        Xb, w_q, 16384, 1024, 1024, Ab, nullptr, nullptr, nullptr, nullptr, 0);
    // 9) kvc = encoder_out @ w_kv  (32768 x 2048 x 1024, f32 A)
    gemm_kernel<EPI_BF16, true><<<dim3(8, 256), 256, 0, stream>>>(
        enc, w_kv, 32768, 2048, 1024, BIG, nullptr, nullptr, nullptr, nullptr, 0);
    // 10) cross-attention (encoder mask) -> Xb
    attn_kernel<1><<<1024, 256, 0, stream>>>(
        Ab, BIG, BIG + 1024, 262144L, 1048576L, 1024, 2048, 512,
        encMask, Xb, 262144L, 1024, 16);
    // 11) x = x + ac @ w_o
    gemm_kernel<EPI_RES, false><<<dim3(4, 128), 256, 0, stream>>>(
        Xb, w_o, 16384, 1024, 1024, nullptr, out, out, nullptr, nullptr, 0);
    // 12) h = LN2(x)*(1+sc_mlp)+sh_mlp
    ln_kernel<<<16384, 256, 0, stream>>>(out, norm2w, mod, SH_MLP, SC_MLP, Ab);
    // 13) m1 = gelu(h @ w_mlp1 + b1)
    gemm_kernel<EPI_BIAS_GELU, false><<<dim3(16, 128), 256, 0, stream>>>(
        Ab, w_m1, 16384, 4096, 1024, BIG, nullptr, nullptr, b_m1, nullptr, 0);
    // 14) out = x + g_mlp * (m1 @ w_mlp2 + b2)
    gemm_kernel<EPI_RES_GATE_BIAS, false><<<dim3(4, 128), 256, 0, stream>>>(
        BIG, w_m2, 16384, 1024, 4096, nullptr, out, out, b_m2, mod, G_MLP);
}
